// MultiHead_43885975830735
// MI455X (gfx1250) — matrix operand order repaired and checked by reading, not compiled, not run on hardware
//
#include <hip/hip_runtime.h>

// ---------------------------------------------------------------------------
// MultiHead attention for MI455X (gfx1250, wave32, WMMA bf16 path).
//   B=4, I=2048, E=1024, H=16, D=64
// Kernel 0: one-shot f32 -> bf16 conversion of x and W into workspace.
// Kernel 1: qkv = x @ W^T  (bf16 WMMA 16x16x32, f32 accumulate, 16x64/wave)
//           -> Q [B,H,I,D] (pre-scaled by D^-0.5), K [B,H,I,D], V^T [B,H,D,I]
// Kernel 2: flash attention, 32 keys per step, online softmax, WMMA for
//           Q@K^T and P@V, P re-laid via per-wave LDS + s_wait_dscnt.
// All fragment loads are 16-byte vector loads (no per-element packing VALU).
// ---------------------------------------------------------------------------

typedef __attribute__((ext_vector_type(16))) __bf16 v16bf;
typedef __attribute__((ext_vector_type(8)))  __bf16 v8bf;
typedef __attribute__((ext_vector_type(8)))  float  v8f;

#define B_  4
#define I_  2048
#define E_  1024
#define H_  16
#define D_  64

__device__ __forceinline__ v16bf concat8(v8bf lo, v8bf hi) {
  return __builtin_shufflevector(lo, hi, 0, 1, 2, 3, 4, 5, 6, 7,
                                         8, 9, 10, 11, 12, 13, 14, 15);
}

__device__ __forceinline__ v8f wmma_bf16(v16bf a, v16bf b, v8f c) {
  return __builtin_amdgcn_wmma_f32_16x16x32_bf16(
      /*neg_a=*/false, a, /*neg_b=*/false, b,
      /*c_mod=*/(short)0, c, /*reuse_a=*/false, /*reuse_b=*/false);
}

// ---------------------------------------------------------------------------
// Kernel 0: one-shot bf16 conversion of x (8.4M) and W (3.1M).
// Each thread converts 8 contiguous floats -> one 16B bf16 store.
// ---------------------------------------------------------------------------
__global__ __launch_bounds__(256)
void convert_kernel(const float* __restrict__ x, const float* __restrict__ w,
                    __bf16* __restrict__ xb, __bf16* __restrict__ wb) {
  const size_t xThreads = (size_t)B_ * I_ * E_ / 8;   // 1,048,576
  const size_t tid = (size_t)blockIdx.x * 256 + threadIdx.x;
  const float* src;
  __bf16* dst;
  size_t base;
  if (tid < xThreads) { src = x; dst = xb; base = tid * 8; }
  else                { src = w; dst = wb; base = (tid - xThreads) * 8; }
  const float4 a = *(const float4*)(src + base);
  const float4 b = *(const float4*)(src + base + 4);
  v8bf r;
  r[0] = (__bf16)a.x; r[1] = (__bf16)a.y; r[2] = (__bf16)a.z; r[3] = (__bf16)a.w;
  r[4] = (__bf16)b.x; r[5] = (__bf16)b.y; r[6] = (__bf16)b.z; r[7] = (__bf16)b.w;
  *(v8bf*)(dst + base) = r;
}

// ---------------------------------------------------------------------------
// Kernel 1: QKV projection. One wave owns a 16x64 strip (4 WMMA tiles sharing
// one A fragment per 32-deep K step) of the [8192, 3072] output.
// ---------------------------------------------------------------------------
__global__ __launch_bounds__(256)
void qkv_gemm_kernel(const __bf16* __restrict__ xb, const __bf16* __restrict__ wb,
                     __bf16* __restrict__ Qw, __bf16* __restrict__ Kw,
                     __bf16* __restrict__ Vt) {
  constexpr int TN4 = (3 * E_) / 64;        // 48 strips along N
  const int lane = threadIdx.x & 31;
  const int wave = threadIdx.x >> 5;
  const int ln   = lane & 15;
  const int hi   = lane >> 4;

  const int tile = blockIdx.x * 8 + wave;   // [0, 512*48)
  const int tn4 = tile % TN4;
  const int tm  = tile / TN4;

  // A fragment: row tm*16+ln, K pattern 16hi + 0..15
  const __bf16* ap = xb + (size_t)(tm * 16 + ln) * E_ + 16 * hi;
  // B fragment j: row (W row) tn4*64 + j*16 + ln, K pattern 16hi + 0..15
  const __bf16* bp = wb + (size_t)(tn4 * 64 + ln) * E_ + 16 * hi;

  v8f c[4] = {};
#pragma unroll 2
  for (int k0 = 0; k0 < E_; k0 += 32) {
    const v16bf a = concat8(*(const v8bf*)(ap + k0),
                            *(const v8bf*)(ap + k0 + 8));
    __builtin_prefetch(ap + k0 + 128, 0, 1);   // global_prefetch_b8
#pragma unroll
    for (int j = 0; j < 4; ++j) {
      const __bf16* bj = bp + (size_t)j * 16 * E_ + k0;
      const v16bf b = concat8(*(const v8bf*)bj, *(const v8bf*)(bj + 8));
      c[j] = wmma_bf16(a, b, c[j]);
    }
  }

  // Store: lane holds (M = r + 8*hi, N = j*16 + ln) for r = 0..7.
#pragma unroll
  for (int j = 0; j < 4; ++j) {
    const int nCol = tn4 * 64 + j * 16 + ln;
#pragma unroll
    for (int r = 0; r < 8; ++r) {
      const int gm   = tm * 16 + r + 8 * hi;   // row in [0, B*I)
      const int bIdx = gm >> 11;               // / I_
      const int iIdx = gm & (I_ - 1);
      const float v  = c[j][r];
      if (nCol < E_) {                         // Q, fold in D^-0.5 = 0.125
        const int h = nCol >> 6, d = nCol & 63;
        Qw[(((size_t)(bIdx * H_ + h)) * I_ + iIdx) * D_ + d] = (__bf16)(v * 0.125f);
      } else if (nCol < 2 * E_) {              // K
        const int n2 = nCol - E_;
        const int h = n2 >> 6, d = n2 & 63;
        Kw[(((size_t)(bIdx * H_ + h)) * I_ + iIdx) * D_ + d] = (__bf16)v;
      } else {                                 // V stored transposed [B,H,D,I]
        const int n2 = nCol - 2 * E_;
        const int h = n2 >> 6, d = n2 & 63;
        Vt[(((size_t)(bIdx * H_ + h)) * D_ + d) * I_ + iIdx] = (__bf16)v;
      }
    }
  }
}

// ---------------------------------------------------------------------------
// Kernel 2: causal flash attention. One wave per 16-query tile of one (b,h).
// ---------------------------------------------------------------------------
__global__ __launch_bounds__(256)
void attn_kernel(const __bf16* __restrict__ Qw, const __bf16* __restrict__ Kw,
                 const __bf16* __restrict__ Vt, float* __restrict__ out) {
  __shared__ __attribute__((aligned(16))) __bf16 pbuf[8 * 16 * 32];
  const int lane = threadIdx.x & 31;
  const int wave = threadIdx.x >> 5;
  const int ln   = lane & 15;
  const int hi   = lane >> 4;
  __bf16* pl = pbuf + wave * 512;

  const int wid   = blockIdx.x * 8 + wave;       // [0, B*H*I/16)
  const int bh    = wid >> 7;                    // / (I/16)
  const int qtile = wid & 127;
  const int qbase = qtile * 16;

  const __bf16* Qb = Qw + (size_t)bh * I_ * D_;
  const __bf16* Kb = Kw + (size_t)bh * I_ * D_;
  const __bf16* Vb = Vt + (size_t)bh * D_ * I_;

  // Q as two A fragments (K-split of D=64 into 2x32), resident all loop long.
  v16bf qa[2];
  {
    const __bf16* qp = Qb + (size_t)(qbase + ln) * D_ + 16 * hi;
#pragma unroll
    for (int s = 0; s < 2; ++s)
      qa[s] = concat8(*(const v8bf*)(qp + s * 32),
                      *(const v8bf*)(qp + s * 32 + 8));
  }

  float mi[8], li[8];
  v8f o[4] = {};
#pragma unroll
  for (int r = 0; r < 8; ++r) { mi[r] = -3.0e38f; li[r] = 0.0f; }

  const int nk = (qtile >> 1) + 1;               // 32-key blocks (causal)
  for (int kb = 0; kb < nk; ++kb) {
    const int kbase = kb * 32;

    // ---- scores: two 16x16 tiles (keys kbase..+15 / +16..+31) ----
    v8f s0 = {}, s1 = {};
#pragma unroll
    for (int s = 0; s < 2; ++s) {
      const __bf16* kp0 = Kb + (size_t)(kbase + ln) * D_ + s * 32 + 16 * hi;
      const __bf16* kp1 = kp0 + 16 * D_;
      const v16bf k0f = concat8(*(const v8bf*)kp0, *(const v8bf*)(kp0 + 8));
      const v16bf k1f = concat8(*(const v8bf*)kp1, *(const v8bf*)(kp1 + 8));
      s0 = wmma_bf16(qa[s], k0f, s0);
      s1 = wmma_bf16(qa[s], k1f, s1);
    }

    // ---- causal mask only on the diagonal block ----
    if (kb == nk - 1) {
#pragma unroll
      for (int r = 0; r < 8; ++r) {
        const int qg = qbase + r + 8 * hi;
        if (kbase + ln > qg)      s0[r] = -1.0e30f;
        if (kbase + 16 + ln > qg) s1[r] = -1.0e30f;
      }
    }

    // ---- online softmax (rows live across 16-lane half-groups) ----
#pragma unroll
    for (int r = 0; r < 8; ++r) {
      float rm = fmaxf(s0[r], s1[r]);
#pragma unroll
      for (int off = 8; off; off >>= 1) rm = fmaxf(rm, __shfl_xor(rm, off, 16));
      const float mnew = fmaxf(mi[r], rm);
      const float corr = __expf(mi[r] - mnew);
      const float p0   = __expf(s0[r] - mnew);
      const float p1   = __expf(s1[r] - mnew);
      float rs = p0 + p1;
#pragma unroll
      for (int off = 8; off; off >>= 1) rs += __shfl_xor(rs, off, 16);
      li[r] = li[r] * corr + rs;
      mi[r] = mnew;
#pragma unroll
      for (int d = 0; d < 4; ++d) o[d][r] = o[d][r] * corr;
      const int row = r + 8 * hi;                 // C-layout row of this lane
      pl[row * 32 + ln]      = (__bf16)p0;
      pl[row * 32 + 16 + ln] = (__bf16)p1;
    }

    // ---- wave-local LDS fence: C-layout -> A-layout relayout of P ----
    __builtin_amdgcn_wave_barrier();
    asm volatile("s_wait_dscnt 0" ::: "memory");
    __builtin_amdgcn_wave_barrier();

    const __bf16* pp = pl + ln * 32 + 16 * hi;
    const v16bf pa = concat8(*(const v8bf*)pp, *(const v8bf*)(pp + 8));

    // ---- O += P @ V  (V^T layout -> contiguous B fragments) ----
#pragma unroll
    for (int d = 0; d < 4; ++d) {
      const __bf16* vp = Vb + (size_t)(d * 16 + ln) * I_ + kbase + 16 * hi;
      const v16bf vf = concat8(*(const v8bf*)vp, *(const v8bf*)(vp + 8));
      o[d] = wmma_bf16(pa, vf, o[d]);
    }
  }

  // ---- epilogue: normalize and scatter to [B, I, E] fp32 ----
  const int bB = bh >> 4, hh = bh & 15;
#pragma unroll
  for (int r = 0; r < 8; ++r) {
    const int row   = qbase + r + 8 * hi;
    const float inv = 1.0f / li[r];
#pragma unroll
    for (int d = 0; d < 4; ++d) {
      out[((size_t)(bB * I_ + row)) * (H_ * D_) + hh * D_ + d * 16 + ln] =
          o[d][r] * inv;
    }
  }
}

// ---------------------------------------------------------------------------
extern "C" void kernel_launch(void* const* d_in, const int* in_sizes, int n_in,
                              void* d_out, int out_size, void* d_ws, size_t ws_size,
                              hipStream_t stream) {
  (void)in_sizes; (void)n_in; (void)out_size; (void)ws_size;
  const float* x = (const float*)d_in[0];       // [B, I, E] fp32
  const float* w = (const float*)d_in[1];       // [3E, E]   fp32
  float* out     = (float*)d_out;               // [B, I, E] fp32

  // Workspace layout (bf16 elements): xb | wb | Q | K | V^T  (~73 MB)
  const size_t xN = (size_t)B_ * I_ * E_;       // 8,388,608
  const size_t wN = (size_t)3 * E_ * E_;        // 3,145,728
  const size_t qN = (size_t)B_ * H_ * I_ * D_;  // 8,388,608
  __bf16* xb = (__bf16*)d_ws;
  __bf16* wb = xb + xN;
  __bf16* Qw = wb + wN;
  __bf16* Kw = Qw + qN;
  __bf16* Vt = Kw + qN;

  // (8.4M + 3.1M) / 8 elems per thread / 256 threads = 5632 blocks
  convert_kernel<<<5632, 256, 0, stream>>>(x, w, xb, wb);
  // 512 * 48 strips / 8 waves per block
  qkv_gemm_kernel<<<3072, 256, 0, stream>>>(xb, wb, Qw, Kw, Vt);
  // 8192 query tiles / 8 waves per block
  attn_kernel<<<1024, 256, 0, stream>>>(Qw, Kw, Vt, out);
}
